// BlobStore_22402549416336
// MI455X (gfx1250) — matrix operand order repaired and checked by reading, not compiled, not run on hardware
//
#include <hip/hip_runtime.h>
#include <hip/hip_bf16.h>

typedef __attribute__((ext_vector_type(16))) _Float16 v16h;
typedef __attribute__((ext_vector_type(8)))  _Float16 v8h;
typedef __attribute__((ext_vector_type(8)))  float    v8f;

#define DS 64
#define DF 256
#define KSEL 16
#define NSPLIT 8
#define TMAXC 0.3f
#define CAND_CAP 4096
#define NEG_INF (-1.0e30f)

__global__ __launch_bounds__(256) void prep_blobs(
    const float* __restrict__ mu, const float* __restrict__ log_var,
    const float* __restrict__ raw_alpha, const float* __restrict__ log_tau,
    _Float16* __restrict__ bmat, float* __restrict__ lct, int N)
{
  const int t  = threadIdx.x;
  const int g  = t >> 6;
  const int d  = t & 63;
  const int nb = blockIdx.x * 4 + g;
  const bool ok = nb < N;

  float m = 0.f, iv = 0.f;
  if (ok) {
    m  = mu[(size_t)nb * DS + d];
    iv = __expf(-log_var[(size_t)nb * DS + d]);
    bmat[(size_t)nb * 128 + d]      = (_Float16)iv;
    bmat[(size_t)nb * 128 + 64 + d] = (_Float16)(-2.0f * m * iv);
  }
  __shared__ float red[4][64];
  red[g][d] = m * m * iv;
  __syncthreads();
  #pragma unroll
  for (int s = 32; s > 0; s >>= 1) {
    if (d < s) red[g][d] += red[g][d + s];
    __syncthreads();
  }
  if (ok && d == 0) {
    float tau   = __expf(log_tau[0]);
    float alpha = 1.0f / (1.0f + __expf(-raw_alpha[nb]));
    lct[nb] = __log2f(alpha) - 0.72134752044448170f * red[g][0] / tau;
  }
}

__global__ __launch_bounds__(256) void prep_queries(
    const float* __restrict__ q, _Float16* __restrict__ amat, int B)
{
  int i = blockIdx.x * 256 + threadIdx.x;
  if (i >= B * DS) return;
  int b = i >> 6, d = i & 63;
  float v = q[(size_t)b * DS + d];
  amat[(size_t)b * 128 + d]      = (_Float16)(v * v);
  amat[(size_t)b * 128 + 64 + d] = (_Float16)v;
}

__global__ __launch_bounds__(256) void score_topk(
    const _Float16* __restrict__ amat, const _Float16* __restrict__ bmat,
    const float* __restrict__ lct, const float* __restrict__ log_tau,
    float* __restrict__ pscore, int* __restrict__ pidx, int N)
{
  __shared__ float ts[16][16];
  __shared__ int   ti[16][16];
  __shared__ float thr[16];
  __shared__ float cand_s[CAND_CAP];
  __shared__ int   cand_m[CAND_CAP];
  __shared__ int   count;

  const int t    = threadIdx.x;
  const int lane = t & 31;
  const int wave = t >> 5;
  const int r0   = blockIdx.x * 16;
  const int nPer = N / NSPLIT;
  const int nbase = blockIdx.y * nPer;

  ((float*)ts)[t] = NEG_INF;
  ((int*)ti)[t]   = 0;
  if (t < 16) thr[t] = NEG_INF;
  if (t == 0) count = 0;
  __syncthreads();

  const float sc = -0.72134752044448170f / __expf(log_tau[0]); // -0.5/(tau*ln2)

  const int m     = lane & 15;
  const int hi    = (lane >> 4) * 8;
  const int khalf = (lane >> 4) * 16;
  v16h afrag[4];
  #pragma unroll
  for (int kc = 0; kc < 4; ++kc) {
    const _Float16* ap = amat + (size_t)(r0 + m) * 128 + kc * 32 + khalf;
    v8h lo  = *(const v8h*)(ap);
    v8h hi8 = *(const v8h*)(ap + 8);
    #pragma unroll
    for (int j = 0; j < 8; ++j) { afrag[kc][j] = lo[j]; afrag[kc][8 + j] = hi8[j]; }
  }

  float thr_reg[8];
  #pragma unroll
  for (int r = 0; r < 8; ++r) thr_reg[r] = NEG_INF;

  const int iters = nPer / 256;
  for (int it = 0; it < iters; ++it) {
    const int n0 = nbase + (it * 16 + wave * 2) * 16 + (lane & 15);
    const int n1 = n0 + 16;
    const _Float16* bp0 = bmat + (size_t)n0 * 128 + khalf;
    const _Float16* bp1 = bmat + (size_t)n1 * 128 + khalf;
    __builtin_prefetch(bp0 + 256 * 128, 0, 1);

    v16h b00 = *(const v16h*)(bp0);
    v16h b01 = *(const v16h*)(bp0 + 32);
    v16h b02 = *(const v16h*)(bp0 + 64);
    v16h b03 = *(const v16h*)(bp0 + 96);
    v16h b10 = *(const v16h*)(bp1);
    v16h b11 = *(const v16h*)(bp1 + 32);
    v16h b12 = *(const v16h*)(bp1 + 64);
    v16h b13 = *(const v16h*)(bp1 + 96);
    const float c0 = lct[n0];
    const float c1 = lct[n1];

    v8f acc0 = {}, acc1 = {};
    acc0 = __builtin_amdgcn_wmma_f32_16x16x32_f16(false, afrag[0], false, b00, (short)0, acc0, false, false);
    acc1 = __builtin_amdgcn_wmma_f32_16x16x32_f16(false, afrag[0], false, b10, (short)0, acc1, false, false);
    acc0 = __builtin_amdgcn_wmma_f32_16x16x32_f16(false, afrag[1], false, b01, (short)0, acc0, false, false);
    acc1 = __builtin_amdgcn_wmma_f32_16x16x32_f16(false, afrag[1], false, b11, (short)0, acc1, false, false);
    acc0 = __builtin_amdgcn_wmma_f32_16x16x32_f16(false, afrag[2], false, b02, (short)0, acc0, false, false);
    acc1 = __builtin_amdgcn_wmma_f32_16x16x32_f16(false, afrag[2], false, b12, (short)0, acc1, false, false);
    acc0 = __builtin_amdgcn_wmma_f32_16x16x32_f16(false, afrag[3], false, b03, (short)0, acc0, false, false);
    acc1 = __builtin_amdgcn_wmma_f32_16x16x32_f16(false, afrag[3], false, b13, (short)0, acc1, false, false);

    #pragma unroll
    for (int r = 0; r < 8; ++r) {
      float ls0 = fmaf(acc0[r], sc, c0);
      if (ls0 > thr_reg[r]) {
        int pos = atomicAdd(&count, 1);
        if (pos < CAND_CAP) { cand_s[pos] = ls0; cand_m[pos] = (n0 << 4) | (r + hi); }
      }
      float ls1 = fmaf(acc1[r], sc, c1);
      if (ls1 > thr_reg[r]) {
        int pos = atomicAdd(&count, 1);
        if (pos < CAND_CAP) { cand_s[pos] = ls1; cand_m[pos] = (n1 << 4) | (r + hi); }
      }
    }

    if ((it == 0) || ((it & 3) == 3) || (it == iters - 1)) {
      __syncthreads();
      if (t < 16) {
        int cnt = count; if (cnt > CAND_CAP) cnt = CAND_CAP;
        for (int i = 0; i < cnt; ++i) {
          int meta = cand_m[i];
          if ((meta & 15) != t) continue;
          float s = cand_s[i];
          if (s <= ts[t][0]) continue;
          int idx = meta >> 4;
          int p = 0;
          while (p < 15 && ts[t][p + 1] < s) { ts[t][p] = ts[t][p + 1]; ti[t][p] = ti[t][p + 1]; ++p; }
          ts[t][p] = s; ti[t][p] = idx;
        }
        thr[t] = ts[t][0];
        if (t == 0) count = 0;
      }
      __syncthreads();
      #pragma unroll
      for (int r = 0; r < 8; ++r) thr_reg[r] = thr[hi + r];
    }
  }

  if (t < 16) {
    size_t base = ((size_t)(r0 + t) * NSPLIT + blockIdx.y) * KSEL;
    for (int j = 0; j < KSEL; ++j) {
      pscore[base + j] = ts[t][15 - j];
      pidx[base + j]   = ti[t][15 - j];
    }
  }
}

__global__ __launch_bounds__(256) void finalize(
    const float* __restrict__ query, const float* __restrict__ mu,
    const float* __restrict__ log_var, const float* __restrict__ raw_alpha,
    const float* __restrict__ features, const float* __restrict__ log_tau,
    const float* __restrict__ pscore, const int* __restrict__ pidx,
    float* __restrict__ out, int B)
{
  __shared__ float cs[NSPLIT * KSEL];
  __shared__ int   ci[NSPLIT * KSEL];
  __shared__ int   sel[KSEL];
  __shared__ float eff[KSEL];
  __shared__ float w[KSEL];
  __shared__ float tres;

  const int b = blockIdx.x;
  const int t = threadIdx.x;
  const int M = NSPLIT * KSEL;
  if (t < M) { cs[t] = pscore[(size_t)b * M + t]; ci[t] = pidx[(size_t)b * M + t]; }
  __syncthreads();

  if (t == 0) {
    for (int j = 0; j < KSEL; ++j) {
      int best = 0; float bs = cs[0];
      for (int i = 1; i < M; ++i) if (cs[i] > bs) { bs = cs[i]; best = i; }
      sel[j] = ci[best]; cs[best] = -2.0e30f;
    }
  }
  __syncthreads();

  const float tau = __expf(log_tau[0]);
  if (t < KSEL) {
    int idx = sel[t];
    float mah = 0.f;
    for (int d = 0; d < DS; ++d) {
      float diff = query[(size_t)b * DS + d] - mu[(size_t)idx * DS + d];
      mah += diff * diff * __expf(-log_var[(size_t)idx * DS + d]);
    }
    float Kv    = __expf(-0.5f * mah / tau);
    float alpha = 1.0f / (1.0f + __expf(-raw_alpha[idx]));
    float e     = alpha * Kv;
    float cap   = TMAXC / (float)KSEL;
    eff[t] = e < cap ? e : cap;
  }
  __syncthreads();

  if (t == 0) {
    float logT = 0.f;
    for (int j = 0; j < KSEL; ++j) {
      float e = eff[j];
      w[j] = e * __expf(logT);
      float ec = e < (1.0f - 1e-6f) ? e : (1.0f - 1e-6f);
      logT += log1pf(-ec);
    }
    tres = __expf(logT);
  }
  __syncthreads();

  float acc = 0.f;
  #pragma unroll 4
  for (int j = 0; j < KSEL; ++j)
    acc += w[j] * features[(size_t)sel[j] * DF + t];
  out[(size_t)b * DF + t] = acc;
  if (t == 0) out[(size_t)B * DF + b] = tres;
}

extern "C" void kernel_launch(void* const* d_in, const int* in_sizes, int n_in,
                              void* d_out, int out_size, void* d_ws, size_t ws_size,
                              hipStream_t stream)
{
  const float* query     = (const float*)d_in[0];
  const float* mu        = (const float*)d_in[1];
  const float* log_var   = (const float*)d_in[2];
  const float* raw_alpha = (const float*)d_in[3];
  const float* features  = (const float*)d_in[4];
  const float* log_tau   = (const float*)d_in[5];
  const int B = in_sizes[0] / DS;
  const int N = in_sizes[1] / DS;

  char*  ws  = (char*)d_ws;
  size_t off = 0;
  auto alloc = [&](size_t bytes) -> void* {
    void* p = ws + off;
    off = (off + bytes + 255) & ~(size_t)255;
    return p;
  };
  _Float16* bmat = (_Float16*)alloc((size_t)N * 128 * sizeof(_Float16));
  _Float16* amat = (_Float16*)alloc((size_t)B * 128 * sizeof(_Float16));
  float*    lct  = (float*)   alloc((size_t)N * sizeof(float));
  float*    psc  = (float*)   alloc((size_t)B * NSPLIT * KSEL * sizeof(float));
  int*      pix  = (int*)     alloc((size_t)B * NSPLIT * KSEL * sizeof(int));

  prep_blobs  <<<(N + 3) / 4, 256, 0, stream>>>(mu, log_var, raw_alpha, log_tau, bmat, lct, N);
  prep_queries<<<(B * DS + 255) / 256, 256, 0, stream>>>(query, amat, B);
  score_topk  <<<dim3(B / 16, NSPLIT), 256, 0, stream>>>(amat, bmat, lct, log_tau, psc, pix, N);
  finalize    <<<B, DF, 0, stream>>>(query, mu, log_var, raw_alpha, features, log_tau, psc, pix,
                                     (float*)d_out, B);
}
